// PMGTEncoder_70325794505399
// MI455X (gfx1250) — compile-verified, matrix operand order checked by reading, not run on hardware
//
#include <hip/hip_runtime.h>
#include <hip/hip_bf16.h>
#include <math.h>

typedef __bf16 bf16;
typedef __attribute__((ext_vector_type(16))) __bf16 v16bf;
typedef __attribute__((ext_vector_type(8)))  __bf16 v8bf;
typedef __attribute__((ext_vector_type(8)))  float  v8f;

// ---------------------------------------------------------------------------
// WMMA helpers (CDNA5: D = A(16x32 bf16) x B(32x16 bf16) + C(16x16 f32))
// ---------------------------------------------------------------------------
__device__ inline v8f wmma_bf16(v16bf a, v16bf b, v8f c) {
    return __builtin_amdgcn_wmma_f32_16x16x32_bf16(
        /*neg_a=*/false, a, /*neg_b=*/false, b,
        /*c_mod=*/(short)0, c, /*reuse_a=*/false, /*reuse_b=*/false);
}

// Load one 16x32 bf16 fragment from LDS laid out row-major with leading
// dimension `ld` (bf16 elements). ISA layout: lane m=lane&15 owns row m,
// half hi=lane>>4 selects K-subgroups {8hi..8hi+7} and {16+8hi..16+8hi+7}.
__device__ inline v16bf load_frag(const bf16* __restrict__ p, int ld, int lane) {
    int m = lane & 15, hi = lane >> 4;
    const bf16* q = p + m * ld + 8 * hi;
    v8bf lo = *(const v8bf*)(q);
    v8bf hh = *(const v8bf*)(q + 16);
    v16bf r;
#pragma unroll
    for (int i = 0; i < 8; ++i) { r[i] = lo[i]; r[i + 8] = hh[i]; }
    return r;
}

// ---------------------------------------------------------------------------
// Generic GEMM: out[M,N] = A[M,K](bf16) @ Bw[K,N](f32 weights) + bias
// optional residual add, exact GELU, f32 and/or bf16 outputs.
// Macro tile 128x128, BK=64, 256 threads = 8 waves (4 M x 2 N), wave = 32x64.
// Software-pipelined: double-buffered LDS; next tile's global loads issued
// into registers before the compute phase; one barrier per K-iteration.
// ---------------------------------------------------------------------------
__global__ __launch_bounds__(256) void gemm_wmma(
    const bf16*  __restrict__ A,
    const float* __restrict__ Bw,
    const float* __restrict__ bias,
    const float* __restrict__ resid,
    float*       __restrict__ outF,
    bf16*        __restrict__ outB,
    int M, int N, int K, int gelu)
{
    __shared__ __align__(16) bf16 As[2][128 * 72];   // [m][k], ld=72
    __shared__ __align__(16) bf16 Bs[2][128 * 72];   // [n][k], ld=72 (transposed)

    const int tid  = threadIdx.x;
    const int wave = tid >> 5, lane = tid & 31;
    const int bm = blockIdx.y * 128, bn = blockIdx.x * 128;
    const int row_w = (wave >> 1) * 32;   // wave's row origin in macro tile
    const int col_w = (wave & 1) * 64;    // wave's col origin in macro tile

    v8f acc[2][4] = {};

    const int ar = tid >> 1, ac0 = (tid & 1) * 32;     // A staging coords
    const int sn = tid & 127, skh = (tid >> 7) * 32;   // B staging coords

    const bf16*  aptr = A  + (long)(bm + ar) * K + ac0;
    const float* bptr = Bw + (long)skh * N + bn + sn;

    v8bf  areg[4];
    float breg[32];

    // ---- prologue: load tile 0 into regs, park in LDS buffer 0 ----
#pragma unroll
    for (int i = 0; i < 4; ++i) areg[i] = *(const v8bf*)(aptr + 8 * i);
#pragma unroll
    for (int i = 0; i < 32; ++i) breg[i] = bptr[(long)i * N];
    {
        bf16* dst = &As[0][ar * 72 + ac0];
#pragma unroll
        for (int i = 0; i < 4; ++i) *(v8bf*)(dst + 8 * i) = areg[i];
        bf16* bdst = &Bs[0][sn * 72 + skh];
#pragma unroll
        for (int i = 0; i < 32; ++i) bdst[i] = (bf16)breg[i];
    }
    __syncthreads();

    const int nk = K >> 6;
    for (int kt = 0; kt < nk; ++kt) {
        const int buf = kt & 1;

        // ---- issue next tile's global loads (latency hidden by WMMA) ----
        if (kt + 1 < nk) {
            const bf16*  anx = aptr + (kt + 1) * 64;
            const float* bnx = bptr + (long)(kt + 1) * 64 * N;
#pragma unroll
            for (int i = 0; i < 4; ++i) areg[i] = *(const v8bf*)(anx + 8 * i);
#pragma unroll
            for (int i = 0; i < 32; ++i) breg[i] = bnx[(long)i * N];
        }

        // ---- compute from LDS buffer `buf` ----
#pragma unroll
        for (int ks = 0; ks < 64; ks += 32) {
            v16bf af[2];
#pragma unroll
            for (int i = 0; i < 2; ++i)
                af[i] = load_frag(&As[buf][(row_w + 16 * i) * 72 + ks], 72, lane);
#pragma unroll
            for (int j = 0; j < 4; ++j) {
                v16bf bfg = load_frag(&Bs[buf][(col_w + 16 * j) * 72 + ks], 72, lane);
#pragma unroll
                for (int i = 0; i < 2; ++i)
                    acc[i][j] = wmma_bf16(af[i], bfg, acc[i][j]);
            }
        }

        // ---- park next tile in the alternate buffer (last read 1 barrier
        //      ago, so writes cannot race the current compute) ----
        if (kt + 1 < nk) {
            bf16* dst = &As[buf ^ 1][ar * 72 + ac0];
#pragma unroll
            for (int i = 0; i < 4; ++i) *(v8bf*)(dst + 8 * i) = areg[i];
            bf16* bdst = &Bs[buf ^ 1][sn * 72 + skh];
#pragma unroll
            for (int i = 0; i < 32; ++i) bdst[i] = (bf16)breg[i];
        }
        __syncthreads();
    }

    // ---- epilogue ----
    const int hi4 = lane >> 4, ln = lane & 15;
#pragma unroll
    for (int j = 0; j < 4; ++j) {
        int colg = bn + col_w + 16 * j + ln;
        float bv = bias ? bias[colg] : 0.f;
#pragma unroll
        for (int i = 0; i < 2; ++i) {
#pragma unroll
            for (int vr = 0; vr < 8; ++vr) {
                int rowg = bm + row_w + 16 * i + vr + 8 * hi4;
                float v = acc[i][j][vr] + bv;
                long idx = (long)rowg * N + colg;
                if (resid) v += resid[idx];
                if (gelu)  v = 0.5f * v * (1.f + erff(v * 0.70710678118654752f));
                if (outF) outF[idx] = v;
                if (outB) outB[idx] = (bf16)v;
            }
        }
    }
}

// ---------------------------------------------------------------------------
// Attention kernel: flash-style with TWO online softmaxes (cosine branch +
// scaled-QK branch), blended at the end: p = beta*p1 + (1-beta)*p2.
// Block = 128 threads (4 waves), each block: one (b,h) and 64 query rows.
// ---------------------------------------------------------------------------
__device__ inline void score_tiles(const bf16* __restrict__ Aw,
                                   const bf16* __restrict__ Brow,
                                   v8f sc[4], int lane) {
    v16bf a0 = load_frag(Aw + 0, 72, lane);
    v16bf a1 = load_frag(Aw + 32, 72, lane);
#pragma unroll
    for (int t = 0; t < 4; ++t) {
        v16bf b0 = load_frag(Brow + (16 * t) * 72, 72, lane);
        v16bf b1 = load_frag(Brow + (16 * t) * 72 + 32, 72, lane);
        sc[t] = wmma_bf16(a1, b1, wmma_bf16(a0, b0, sc[t]));
    }
}

__device__ inline void online_update(v8f sc[4], float* mst, float* lst,
                                     v8f o[4], bf16* __restrict__ PsW, int lane) {
    int hi4 = lane >> 4, ln = lane & 15;
#pragma unroll
    for (int vr = 0; vr < 8; ++vr) {
        float rmax = -1e30f;
#pragma unroll
        for (int t = 0; t < 4; ++t) rmax = fmaxf(rmax, sc[t][vr]);
#pragma unroll
        for (int m = 1; m < 16; m <<= 1) rmax = fmaxf(rmax, __shfl_xor(rmax, m, 32));
        float mnew = fmaxf(mst[vr], rmax);
        float corr = __expf(mst[vr] - mnew);
        float rsum = 0.f;
#pragma unroll
        for (int t = 0; t < 4; ++t) {
            float p = __expf(sc[t][vr] - mnew);
            rsum += p;
            PsW[(vr + 8 * hi4) * 72 + 16 * t + ln] = (bf16)p;
        }
#pragma unroll
        for (int m = 1; m < 16; m <<= 1) rsum += __shfl_xor(rsum, m, 32);
        lst[vr] = lst[vr] * corr + rsum;
        mst[vr] = mnew;
#pragma unroll
        for (int t = 0; t < 4; ++t) o[t][vr] *= corr;
    }
}

__device__ inline void pv_accum(v8f o[4], const bf16* __restrict__ PsW,
                                const bf16* __restrict__ Vt, int lane) {
    v16bf p0 = load_frag(PsW + 0, 72, lane);
    v16bf p1 = load_frag(PsW + 32, 72, lane);
#pragma unroll
    for (int t = 0; t < 4; ++t) {
        v16bf v0 = load_frag(Vt + (16 * t) * 72, 72, lane);
        v16bf v1 = load_frag(Vt + (16 * t) * 72 + 32, 72, lane);
        o[t] = wmma_bf16(p1, v1, wmma_bf16(p0, v0, o[t]));
    }
}

__global__ __launch_bounds__(128) void attn_kernel(
    const bf16* __restrict__ Qg, const bf16* __restrict__ Kg,
    const bf16* __restrict__ Vg, const bf16* __restrict__ Cg,
    bf16* __restrict__ ctx)
{
    const int S = 512, D = 768;
    const float scale = 0.125f, beta = 0.5f;
    const int q0 = blockIdx.x * 64;
    const int h = blockIdx.y, b = blockIdx.z;
    const int tid = threadIdx.x, wave = tid >> 5, lane = tid & 31;
    const int hi4 = lane >> 4, ln = lane & 15;

    __shared__ __align__(16) bf16 Qs[64 * 72], Cq[64 * 72];
    __shared__ __align__(16) bf16 Ks[64 * 72], Cs[64 * 72], Vt[64 * 72];
    __shared__ __align__(16) bf16 Ps[4 * 16 * 72];
    __shared__ float cnq[64], cnk[64];

    const long base = ((long)b * S) * D + h * 64;

    // stage Q and C-query tiles (64 rows x 64 dims)
    for (int it = tid; it < 512; it += 128) {
        int r = it >> 3, c = (it & 7) * 8;
        long g = base + (long)(q0 + r) * D + c;
        *(v8bf*)&Qs[r * 72 + c] = *(const v8bf*)&Qg[g];
        *(v8bf*)&Cq[r * 72 + c] = *(const v8bf*)&Cg[g];
    }
    __syncthreads();
    if (tid < 64) {
        float s = 0.f;
        for (int d = 0; d < 64; ++d) { float x = (float)Cq[tid * 72 + d]; s += x * x; }
        cnq[tid] = sqrtf(s);
    }
    __syncthreads();

    float m1[8], l1[8], m2[8], l2[8];
    v8f o1[4] = {}, o2[4] = {};
#pragma unroll
    for (int i = 0; i < 8; ++i) { m1[i] = m2[i] = -1e30f; l1[i] = l2[i] = 0.f; }

    const bf16* QsW = &Qs[wave * 16 * 72];
    const bf16* CqW = &Cq[wave * 16 * 72];
    bf16* PsW = &Ps[wave * 16 * 72];

    for (int kt = 0; kt < 8; ++kt) {
        int k0 = kt * 64;
        // stage K, C-key, V (V transposed -> Vt[d][r])
        for (int it = tid; it < 512; it += 128) {
            int r = it >> 3, c = (it & 7) * 8;
            long g = base + (long)(k0 + r) * D + c;
            *(v8bf*)&Ks[r * 72 + c] = *(const v8bf*)&Kg[g];
            *(v8bf*)&Cs[r * 72 + c] = *(const v8bf*)&Cg[g];
            v8bf vv = *(const v8bf*)&Vg[g];
#pragma unroll
            for (int i = 0; i < 8; ++i) Vt[(c + i) * 72 + r] = vv[i];
        }
        __syncthreads();
        if (tid < 64) {
            float s = 0.f;
            for (int d = 0; d < 64; ++d) { float x = (float)Cs[tid * 72 + d]; s += x * x; }
            cnk[tid] = sqrtf(s);
        }
        __syncthreads();

        // ---- branch 1: s1 = 1 - cos(c_l, c_r) + eye ----
        {
            v8f sc[4] = {};
            score_tiles(CqW, Cs, sc, lane);
#pragma unroll
            for (int t = 0; t < 4; ++t) {
                int kl = 16 * t + ln;
                float ck = cnk[kl];
#pragma unroll
                for (int vr = 0; vr < 8; ++vr) {
                    int ql = wave * 16 + vr + 8 * hi4;
                    float dn = cnq[ql] * ck + 1e-20f;
                    float v = 1.f - sc[t][vr] / dn;
                    if (q0 + ql == k0 + kl) v += 1.f;
                    sc[t][vr] = v;
                }
            }
            online_update(sc, m1, l1, o1, PsW, lane);
            pv_accum(o1, PsW, Vt, lane);
        }
        // ---- branch 2: s2 = (q . k) * scale ----
        {
            v8f sc[4] = {};
            score_tiles(QsW, Ks, sc, lane);
#pragma unroll
            for (int t = 0; t < 4; ++t)
#pragma unroll
                for (int vr = 0; vr < 8; ++vr) sc[t][vr] *= scale;
            online_update(sc, m2, l2, o2, PsW, lane);
            pv_accum(o2, PsW, Vt, lane);
        }
        __syncthreads();
    }

    // blend and write ctx (bf16, [B,S,D] head slice)
#pragma unroll
    for (int t = 0; t < 4; ++t) {
        int d = 16 * t + ln;
#pragma unroll
        for (int vr = 0; vr < 8; ++vr) {
            int ql = wave * 16 + vr + 8 * hi4;
            float val = beta * (o1[t][vr] / l1[vr]) + (1.f - beta) * (o2[t][vr] / l2[vr]);
            ctx[base + (long)(q0 + ql) * D + d] = (bf16)val;
        }
    }
}

// ---------------------------------------------------------------------------
// LayerNorm over D=768, one block (256 threads) per row. Emits f32 + bf16.
// ---------------------------------------------------------------------------
__global__ __launch_bounds__(256) void ln_kernel(
    const float* __restrict__ in, const float* __restrict__ gw,
    const float* __restrict__ bw, float* __restrict__ outF,
    bf16* __restrict__ outB)
{
    const int D = 768;
    const long row = blockIdx.x;
    const int tid = threadIdx.x;
    float x0 = in[row * D + tid];
    float x1 = in[row * D + tid + 256];
    float x2 = in[row * D + tid + 512];
    float s = x0 + x1 + x2, s2 = x0 * x0 + x1 * x1 + x2 * x2;
#pragma unroll
    for (int m = 1; m < 32; m <<= 1) { s += __shfl_xor(s, m, 32); s2 += __shfl_xor(s2, m, 32); }
    __shared__ float r1[8], r2[8];
    if ((tid & 31) == 0) { r1[tid >> 5] = s; r2[tid >> 5] = s2; }
    __syncthreads();
    s = 0.f; s2 = 0.f;
#pragma unroll
    for (int i = 0; i < 8; ++i) { s += r1[i]; s2 += r2[i]; }
    float mu   = s * (1.f / 768.f);
    float var  = s2 * (1.f / 768.f) - mu * mu;
    float rstd = rsqrtf(var + 1e-12f);
    float xs[3] = {x0, x1, x2};
#pragma unroll
    for (int j = 0; j < 3; ++j) {
        int col = tid + j * 256;
        float y = (xs[j] - mu) * rstd * gw[col] + bw[col];
        outF[row * D + col] = y;
        if (outB) outB[row * D + col] = (bf16)y;
    }
}

__global__ __launch_bounds__(256) void cvt_kernel(const float* __restrict__ in,
                                                  bf16* __restrict__ out, long n)
{
    long i = (long)blockIdx.x * 256 + threadIdx.x;
    if (i < n) out[i] = (bf16)in[i];
}

// ---------------------------------------------------------------------------
// Host side
// ---------------------------------------------------------------------------
extern "C" void kernel_launch(void* const* d_in, const int* in_sizes, int n_in,
                              void* d_out, int out_size, void* d_ws, size_t ws_size,
                              hipStream_t stream) {
    (void)in_sizes; (void)n_in; (void)out_size; (void)ws_size;
    const int Bn = 16, S = 512, D = 768, F = 3072, L = 4, H = 12;
    const int M = Bn * S;  // 8192

    const float* hidden = (const float*)d_in[0];
    const float* Wq = (const float*)d_in[1];  const float* bq = (const float*)d_in[2];
    const float* Wk = (const float*)d_in[3];  const float* bk = (const float*)d_in[4];
    const float* Wv = (const float*)d_in[5];  const float* bv = (const float*)d_in[6];
    const float* Wc = (const float*)d_in[7];  const float* bc = (const float*)d_in[8];
    const float* Wo = (const float*)d_in[9];  const float* bo = (const float*)d_in[10];
    const float* g1 = (const float*)d_in[11]; const float* b1 = (const float*)d_in[12];
    const float* Wi = (const float*)d_in[13]; const float* bi = (const float*)d_in[14];
    const float* Wo2 = (const float*)d_in[15]; const float* bo2 = (const float*)d_in[16];
    const float* g2 = (const float*)d_in[17]; const float* b2 = (const float*)d_in[18];

    // workspace carve
    char* wp = (char*)d_ws;
    auto take = [&](size_t bytes) -> void* {
        void* r = (void*)wp;
        wp += (bytes + 255) & ~(size_t)255;
        return r;
    };
    bf16* qb      = (bf16*)take((size_t)M * D * 2);
    bf16* kb      = (bf16*)take((size_t)M * D * 2);
    bf16* vb      = (bf16*)take((size_t)M * D * 2);
    bf16* cb      = (bf16*)take((size_t)M * D * 2);
    bf16* ctxb    = (bf16*)take((size_t)M * D * 2);
    bf16* xb16    = (bf16*)take((size_t)M * D * 2);
    bf16* attnb16 = (bf16*)take((size_t)M * D * 2);
    bf16* hb16    = (bf16*)take((size_t)M * F * 2);
    float* pre    = (float*)take((size_t)M * D * 4);
    float* attnf  = (float*)take((size_t)M * D * 4);
    float* xf     = (float*)take((size_t)M * D * 4);
    float* outp   = (float*)d_out;

    // x (layer 0) -> bf16
    {
        long n = (long)M * D;
        cvt_kernel<<<(unsigned)((n + 255) / 256), 256, 0, stream>>>(hidden, xb16, n);
    }

    dim3 gD(D / 128, M / 128);   // N=768 outputs
    dim3 gF(F / 128, M / 128);   // N=3072 outputs
    dim3 gA(S / 64, H, Bn);      // attention

    for (int i = 0; i < L; ++i) {
        const float* xresid = (i == 0) ? hidden : xf;
        long wD = (long)i * D * D, wF = (long)i * D * F;

        // Q,K,V,C projections -> bf16
        gemm_wmma<<<gD, 256, 0, stream>>>(xb16, Wq + wD, bq + (long)i * D, nullptr, nullptr, qb, M, D, D, 0);
        gemm_wmma<<<gD, 256, 0, stream>>>(xb16, Wk + wD, bk + (long)i * D, nullptr, nullptr, kb, M, D, D, 0);
        gemm_wmma<<<gD, 256, 0, stream>>>(xb16, Wv + wD, bv + (long)i * D, nullptr, nullptr, vb, M, D, D, 0);
        gemm_wmma<<<gD, 256, 0, stream>>>(xb16, Wc + wD, bc + (long)i * D, nullptr, nullptr, cb, M, D, D, 0);

        // dual-branch attention -> ctx (bf16)
        attn_kernel<<<gA, 128, 0, stream>>>(qb, kb, vb, cb, ctxb);

        // attn_pre = ctx @ Wo + bo + x ; LN -> attn_out (f32 + bf16)
        gemm_wmma<<<gD, 256, 0, stream>>>(ctxb, Wo + wD, bo + (long)i * D, xresid, pre, nullptr, M, D, D, 0);
        ln_kernel<<<M, 256, 0, stream>>>(pre, g1 + (long)i * D, b1 + (long)i * D, attnf, attnb16);

        // h = gelu(attn_out @ Wi + bi) -> bf16
        gemm_wmma<<<gF, 256, 0, stream>>>(attnb16, Wi + wF, bi + (long)i * F, nullptr, nullptr, hb16, M, F, D, 1);

        // pre2 = h @ Wo2 + bo2 + attn_out ; LN -> x (next layer, or final out)
        gemm_wmma<<<gD, 256, 0, stream>>>(hb16, Wo2 + wF, bo2 + (long)i * D, attnf, pre, nullptr, M, D, F, 0);
        ln_kernel<<<M, 256, 0, stream>>>(pre, g2 + (long)i * D, b2 + (long)i * D,
                                         (i == L - 1) ? outp : xf,
                                         (i == L - 1) ? nullptr : xb16);
    }
}
